// PillarEncoder_50740743635700
// MI455X (gfx1250) — matrix operand order repaired and checked by reading, not compiled, not run on hardware
//
#include <hip/hip_runtime.h>

typedef __attribute__((ext_vector_type(16))) _Float16 v16h;
typedef __attribute__((ext_vector_type(8)))  float    v8f;

#define GH   512
#define GW   512
#define HP   (GH * GW)          // 262144 = 2^18
#define CIN  4
#define HID  64
#define COUT 256
#define NAUG 10
#define X0C  (-51.2f)
#define Y0C  (-51.2f)
#define VSC  (0.2f)

// ---------------------------------------------------------------- zero fill
__global__ void zero_f4(float4* __restrict__ p, long n4) {
  long i = (long)blockIdx.x * blockDim.x + threadIdx.x;
  if (i < n4) p[i] = make_float4(0.f, 0.f, 0.f, 0.f);
}

// ------------------------------------------------- fused MLP + scatter-add
__global__ __launch_bounds__(256) void pillar_mlp_scatter(
    const float* __restrict__ points,
    const float* __restrict__ W1, const float* __restrict__ b1,
    const float* __restrict__ g1, const float* __restrict__ be1,
    const float* __restrict__ m1, const float* __restrict__ v1,
    const float* __restrict__ W2, const float* __restrict__ b2,
    const float* __restrict__ g2, const float* __restrict__ be2,
    const float* __restrict__ m2, const float* __restrict__ v2,
    float* __restrict__ out_sums, float* __restrict__ counts,
    int N, int total_pts)
{
  // W2 pre-swizzled into WMMA B-fragment layout: frag f = t*2+s, 32 lanes,
  // 16 f16 per lane (contiguous, 32B aligned -> 2x ds_load_b128 per frag).
  __shared__ v16h  w2f[16 * 2 * 32];           // 32 KB
  // Layer-1 activations written directly in WMMA A-fragment layout.
  __shared__ v16h  haf[8][2][32];              // 16 KB
  __shared__ float  s1s[HID], t1s[HID];
  __shared__ float2 bn2[COUT];                 // {scale, shift} for layer 2
  __shared__ float  w1s[HID * NAUG];
  __shared__ float  augs[8][16][NAUG];
  __shared__ int    pils[8][16];

  const int tid = threadIdx.x;

  // ---- cooperative preload -------------------------------------------
  // B-fragment element map (ISA 7.12.2, 16-bit B 32x16):
  //   lane: col = lane&15, khalf = lane>>4 ; element i -> K = s*32+(i>>3)*16+khalf*8+(i&7)
  for (int i = tid; i < 16 * 2 * 32 * 16; i += 256) {
    const int e  = i & 15;
    const int ln = (i >> 4) & 31;
    const int f  = i >> 9;                     // t*2 + s
    const int s  = f & 1, t = f >> 1;
    const int K  = s * 32 + (e >> 3) * 16 + (ln >> 4) * 8 + (e & 7);
    const int C  = t * 16 + (ln & 15);
    ((_Float16*)w2f)[i] = (_Float16)W2[C * HID + K];
  }
  if (tid < HID) {
    float s = g1[tid] * rsqrtf(v1[tid] + 1e-3f);
    s1s[tid] = s;
    t1s[tid] = (b1[tid] - m1[tid]) * s + be1[tid];
  }
  {
    float s = g2[tid] * rsqrtf(v2[tid] + 1e-3f);
    bn2[tid] = make_float2(s, (b2[tid] - m2[tid]) * s + be2[tid]);
  }
  for (int i = tid; i < HID * NAUG; i += 256) w1s[i] = W1[i];
  __syncthreads();

  const int wave = tid >> 5, lane = tid & 31;
  const int half = lane >> 4, col = lane & 15;

  const long tile = (long)blockIdx.x * 8 + wave;
  const long gp0  = tile * 16;
  const bool tile_ok = (gp0 < (long)total_pts);
  int  b  = (int)(gp0 / N); if (b > 1) b = 1;
  const int n0 = (int)(gp0 - (long)b * N);

  // ---- per-point featurization (lanes 0..15 own one point each) -------
  if (lane < 16) {
    const int n = n0 + lane;
    int pidx = -1;
    float a[NAUG] = {0.f,0.f,0.f,0.f,0.f,0.f,0.f,0.f,0.f,0.f};
    if (tile_ok && n < N) {
      const float4 pt = ((const float4*)points)[(long)b * N + n];
      if (n + 256 < N)  // warm L2 for upcoming tiles -> global_prefetch_b8
        __builtin_prefetch(&points[((long)b * N + n + 256) * 4], 0, 0);
      const float x = pt.x, y = pt.y;
      const int gx = (int)((x - X0C) / VSC);   // trunc toward zero == .long()
      const int gy = (int)((y - Y0C) / VSC);
      const bool ok = (gx >= 0) & (gx < GW) & (gy >= 0) & (gy < GH);
      const float cx = gx * VSC + X0C + 0.5f * VSC;
      const float cy = gy * VSC + Y0C + 0.5f * VSC;
      a[0] = x; a[1] = y; a[2] = pt.z; a[3] = pt.w;
      a[4] = x - cx; a[5] = y - cy; a[6] = 0.f;
      a[7] = cx; a[8] = cy; a[9] = 0.f;
      if (ok) pidx = gy * GW + gx;
    }
    pils[wave][lane] = pidx;
    #pragma unroll
    for (int f = 0; f < NAUG; ++f) augs[wave][lane][f] = a[f];
    if (pidx >= 0) atomicAdd(&counts[(long)b * HP + pidx], 1.0f);
  }
  __syncthreads();

  // ---- layer 1: scalar FMA (K=10), ReLU, store in A-fragment layout ---
  // A-layout inverse (16-bit A 16x32): k32 = k&31,
  //   half = (k32>>3)&1 ; elem = (k32&7) + ((k32>>4)<<3) ; lane = half*16+p
  for (int idx = lane; idx < 16 * HID; idx += 32) {
    const int p = idx >> 6, k = idx & (HID - 1);
    float dot = 0.f;
    #pragma unroll
    for (int f = 0; f < NAUG; ++f) dot = fmaf(augs[wave][p][f], w1s[k * NAUG + f], dot);
    float h = dot * s1s[k] + t1s[k];
    h = h > 0.f ? h : 0.f;
    const int s   = k >> 5;
    const int k32 = k & 31;
    const int hh  = (k32 >> 3) & 1;
    const int el  = (k32 & 7) + ((k32 >> 4) << 3);
    ((_Float16*)&haf[wave][s][hh * 16 + p])[el] = (_Float16)h;
  }
  __syncthreads();

  // ---- A fragments: contiguous 32B per lane -> ds_load_b128 pairs -----
  const v16h a0 = haf[wave][0][lane];
  const v16h a1 = haf[wave][1][lane];

  int prow[8];
  #pragma unroll
  for (int r = 0; r < 8; ++r) prow[r] = pils[wave][r + 8 * half];

  float* const outb = out_sums + (long)b * COUT * HP;

  // ---- layer 2: 16 N-tiles x 2 WMMA (v_wmma_f32_16x16x32_f16) ---------
  #pragma unroll 2
  for (int t = 0; t < 16; ++t) {
    const v16h bf0 = w2f[(2 * t + 0) * 32 + lane];
    const v16h bf1 = w2f[(2 * t + 1) * 32 + lane];
    v8f acc = {};
    acc = __builtin_amdgcn_wmma_f32_16x16x32_f16(false, a0, false, bf0,
                                                 (short)0, acc, false, false);
    acc = __builtin_amdgcn_wmma_f32_16x16x32_f16(false, a1, false, bf1,
                                                 (short)0, acc, false, false);
    const int    ch = t * 16 + col;
    const float2 ss = bn2[ch];
    float* const outc = outb + (long)ch * HP;
    #pragma unroll
    for (int r = 0; r < 8; ++r) {
      if (prow[r] >= 0) {
        float v = acc[r] * ss.x + ss.y;      // fused BN
        v = v > 0.f ? v : 0.f;               // ReLU
        v = v < 100.f ? v : 100.f;           // clip (lower bound moot post-ReLU)
        atomicAdd(outc + prow[r], v);        // global_atomic_add_f32
      }
    }
  }
}

// ------------------------------------------------- sums -> mean (float4)
__global__ void finalize_div(float4* __restrict__ out,
                             const float4* __restrict__ counts4, long n4) {
  long i = (long)blockIdx.x * blockDim.x + threadIdx.x;
  if (i >= n4) return;
  const long base = i * 4;
  const long p  = base & (long)(HP - 1);     // HP = 2^18
  const long bc = base >> 18;
  const long b  = bc >> 8;                   // COUT = 2^8
  const float4 cn = counts4[((long)b * HP + p) >> 2];
  float4 v = out[i];
  v.x /= (cn.x + 1e-6f);
  v.y /= (cn.y + 1e-6f);
  v.z /= (cn.z + 1e-6f);
  v.w /= (cn.w + 1e-6f);
  out[i] = v;
}

extern "C" void kernel_launch(void* const* d_in, const int* in_sizes, int n_in,
                              void* d_out, int out_size, void* d_ws, size_t ws_size,
                              hipStream_t stream) {
  const float* points = (const float*)d_in[0];
  const float* W1  = (const float*)d_in[1];
  const float* b1  = (const float*)d_in[2];
  const float* g1  = (const float*)d_in[3];
  const float* be1 = (const float*)d_in[4];
  const float* m1  = (const float*)d_in[5];
  const float* v1  = (const float*)d_in[6];
  const float* W2  = (const float*)d_in[7];
  const float* b2  = (const float*)d_in[8];
  const float* g2  = (const float*)d_in[9];
  const float* be2 = (const float*)d_in[10];
  const float* m2  = (const float*)d_in[11];
  const float* v2  = (const float*)d_in[12];

  float* out    = (float*)d_out;
  float* counts = (float*)d_ws;              // 2*HP floats = 2 MB

  const int B = 2;
  const int total_pts = in_sizes[0] / 4;     // B*N
  const int N = total_pts / B;

  const long cnt_n4 = (long)B * HP / 4;
  zero_f4<<<(unsigned)((cnt_n4 + 255) / 256), 256, 0, stream>>>((float4*)counts, cnt_n4);

  const long out_n4 = (long)out_size / 4;
  zero_f4<<<(unsigned)((out_n4 + 255) / 256), 256, 0, stream>>>((float4*)out, out_n4);

  const long ntiles  = ((long)total_pts + 15) / 16;
  const long nblocks = (ntiles + 7) / 8;
  pillar_mlp_scatter<<<(unsigned)nblocks, 256, 0, stream>>>(
      points, W1, b1, g1, be1, m1, v1, W2, b2, g2, be2, m2, v2,
      out, counts, N, total_pts);

  finalize_div<<<(unsigned)((out_n4 + 255) / 256), 256, 0, stream>>>(
      (float4*)out, (const float4*)counts, out_n4);
}
